// HelixNetHybridAttention_76888504533219
// MI455X (gfx1250) — matrix operand order repaired and checked by reading, not compiled, not run on hardware
//
#include <hip/hip_runtime.h>

// ---------------------------------------------------------------------------
// HelixNet hybrid attention for MI455X (gfx1250, wave32, WMMA + TDM)
// B=2, T=2048, HID=1024, NH=8, HD=128, WH=8, WK=64, WV=128, FUSED=6160
// ---------------------------------------------------------------------------

typedef __attribute__((ext_vector_type(8)))  float  v8f;
typedef __attribute__((ext_vector_type(8)))  __bf16 bf16x8;
typedef __attribute__((ext_vector_type(16))) __bf16 bf16x16;
typedef unsigned int u32x4 __attribute__((ext_vector_type(4)));
typedef int          i32x8 __attribute__((ext_vector_type(8)));
typedef int          i32x4 __attribute__((ext_vector_type(4)));

#define B_   2
#define T_   2048
#define HID_ 1024
#define NH_  8
#define HD_  128
#define WH_  8
#define WK_  64
#define WV_  128
#define FUSED_ 6160

// fused column offsets
#define OFF_Q    0
#define OFF_K    1024
#define OFF_V    2048
#define OFF_QW   3072
#define OFF_KW   3584
#define OFF_VW   4096
#define OFF_BETA 5120
#define OFF_DEC  5128
#define OFF_GWB  5136

__device__ __forceinline__ bf16x16 bf_combine(bf16x8 lo, bf16x8 hi) {
  bf16x16 r;
#pragma unroll
  for (int i = 0; i < 8; ++i) { r[i] = lo[i]; r[i + 8] = hi[i]; }
  return r;
}

__device__ __forceinline__ float sigmoidf_(float x) {
  return 1.0f / (1.0f + __expf(-x));
}

// ---------------------------------------------------------------------------
// Tensor Data Mover: 2D tile of bf16 (2-byte) elements, global -> LDS, with
// LDS row padding produced by the TDM pad unit (D# group1 pad fields).
//   tile_d0 elems per row, tile_d1 rows, row stride stride0 elems.
//   pad_i: pad interval code (2^(code+1) DWORDs), pad_a: pad amount code
//   (code+1 DWORDs appended per interval).
// D# layout per CDNA5 ISA; 6-arg builtin:
//   (u32x4 g0, i32x8 g1, i32x4 g2, i32x4 g3, i32x8 g4, i32 cpol)
// ---------------------------------------------------------------------------
__device__ __forceinline__ void tdm_load_2d_bf16(
    unsigned lds_addr, const void* gptr,
    unsigned tensor_d0, unsigned tile_d0, unsigned tile_d1,
    unsigned long long stride0, unsigned pad_i, unsigned pad_a)
{
  const unsigned long long ga = (unsigned long long)(uintptr_t)gptr;
  u32x4 g0;
  g0[0] = 1u;                                        // count=1, user desc
  g0[1] = lds_addr;                                  // lds_addr [63:32]
  g0[2] = (unsigned)(ga & 0xFFFFFFFFu);              // global_addr lo
  g0[3] = (unsigned)((ga >> 32) & 0x01FFFFFFu)       // global_addr hi (56:32)
        | (2u << 30);                                // type = 2 (image)
  const unsigned td1 = 0x200000u;                    // big tensor_dim1 (no OOB clip)
  i32x8 g1;
  g1[0] = (int)((1u << 16)                           // data_size = 1 -> 2 bytes
              | (1u << 20)                           // pad_enable
              | (pad_i << 22) | (pad_a << 25));      // LDS padding
  g1[1] = (int)((tensor_d0 & 0xFFFFu) << 16);        // tensor_dim0[15:0] @ bit16
  g1[2] = (int)((tensor_d0 >> 16) | ((td1 & 0xFFFFu) << 16));
  g1[3] = (int)((td1 >> 16) | (tile_d0 << 16));      // tile_dim0 @ bits 127:112
  g1[4] = (int)(tile_d1 & 0xFFFFu);                  // tile_dim1; tile_dim2 = 0
  g1[5] = (int)(unsigned)(stride0 & 0xFFFFFFFFu);    // tensor_dim0_stride lo
  g1[6] = (int)(unsigned)((stride0 >> 32) & 0xFFFFu);
  g1[7] = 0;                                         // tensor_dim1_stride = 0
  const i32x4 z4 = {0, 0, 0, 0};
  const i32x8 z8 = {0, 0, 0, 0, 0, 0, 0, 0};
  __builtin_amdgcn_tensor_load_to_lds(g0, g1, z4, z4, z8, 0);
}

// ---------------------------------------------------------------------------
// Generic GEMM: C[M,N] = A[M,K] * B[K,N], fp32 in/out, bf16 WMMA compute.
// Block tile 64(M) x 128(N), K step 32. 256 threads = 8 waves in 2x4 grid,
// each wave computes a 32x32 tile = 2x2 v_wmma_f32_16x16x32_bf16.
// ---------------------------------------------------------------------------
#define G_LDA 40   // padded bf16 stride for A tile (64 x 32)
#define G_LDB 40   // padded bf16 stride for B^T tile (128 x 32)

__global__ __launch_bounds__(256) void gemm_bf16_wmma(
    const float* __restrict__ A, const float* __restrict__ Bm,
    float* __restrict__ C, int M, int N, int K)
{
  __shared__ __bf16 As[64 * G_LDA];
  __shared__ __bf16 Bt[128 * G_LDB];

  const int tid  = threadIdx.x;
  const int lane = tid & 31;
  const int wid  = tid >> 5;
  const int sub  = lane >> 4;     // half-wave 0/1
  const int lr   = lane & 15;
  const int wm   = wid >> 2;      // 0..1
  const int wn   = wid & 3;       // 0..3
  const int m0   = blockIdx.y * 64;
  const int n0   = blockIdx.x * 128;
  const bool full_n = (n0 + 128 <= N);   // uniform across block

  v8f acc[2][2] = {};

  for (int k0 = 0; k0 < K; k0 += 32) {
    // ---- stage A tile 64x32 (fp32 -> bf16), row-major ----
    {
      int slot = tid;
#pragma unroll
      for (int it = 0; it < 2; ++it, slot += 256) {
        const int r  = slot >> 3;
        const int c4 = (slot & 7) << 2;
        const float4 v = *(const float4*)(A + (size_t)(m0 + r) * K + k0 + c4);
        As[r * G_LDA + c4 + 0] = (__bf16)v.x;
        As[r * G_LDA + c4 + 1] = (__bf16)v.y;
        As[r * G_LDA + c4 + 2] = (__bf16)v.z;
        As[r * G_LDA + c4 + 3] = (__bf16)v.w;
      }
    }
    // ---- stage B tile 32x128 transposed (fp32 -> bf16), N-major ----
    if (full_n) {
      int slot = tid;
#pragma unroll
      for (int it = 0; it < 4; ++it, slot += 256) {
        const int r  = slot >> 5;          // k row 0..31
        const int c4 = (slot & 31) << 2;   // n col 0..124
        const float4 v = *(const float4*)(Bm + (size_t)(k0 + r) * N + n0 + c4);
        Bt[(c4 + 0) * G_LDB + r] = (__bf16)v.x;
        Bt[(c4 + 1) * G_LDB + r] = (__bf16)v.y;
        Bt[(c4 + 2) * G_LDB + r] = (__bf16)v.z;
        Bt[(c4 + 3) * G_LDB + r] = (__bf16)v.w;
      }
    } else {
      int slot = tid;
#pragma unroll
      for (int it = 0; it < 4; ++it, slot += 256) {
        const int r  = slot >> 5;
        const int c4 = (slot & 31) << 2;
        const float* bp = Bm + (size_t)(k0 + r) * N;
#pragma unroll
        for (int i = 0; i < 4; ++i) {
          const float x = (n0 + c4 + i < N) ? bp[n0 + c4 + i] : 0.f;
          Bt[(c4 + i) * G_LDB + r] = (__bf16)x;
        }
      }
    }
    __syncthreads();

    // prefetch next tiles into cache hierarchy (global_prefetch_b8)
    if (k0 + 32 < K) {
      __builtin_prefetch(A + (size_t)(m0 + (tid >> 2)) * K + (k0 + 32) + ((tid & 3) << 3), 0, 0);
      __builtin_prefetch(Bm + (size_t)(k0 + 32 + (tid >> 3)) * N + n0 + ((tid & 7) << 4), 0, 0);
    }

    // ---- WMMA: one 16x16x32 per output tile per K step ----
#pragma unroll
    for (int tm = 0; tm < 2; ++tm) {
      const int mr = wm * 32 + tm * 16 + lr;
      const bf16x8 alo = *(const bf16x8*)&As[mr * G_LDA + sub * 8];
      const bf16x8 ahi = *(const bf16x8*)&As[mr * G_LDA + sub * 8 + 16];
      const bf16x16 a = bf_combine(alo, ahi);
#pragma unroll
      for (int tn = 0; tn < 2; ++tn) {
        const int nc = wn * 32 + tn * 16 + lr;
        const bf16x8 blo = *(const bf16x8*)&Bt[nc * G_LDB + sub * 8];
        const bf16x8 bhi = *(const bf16x8*)&Bt[nc * G_LDB + sub * 8 + 16];
        const bf16x16 b = bf_combine(blo, bhi);
        acc[tm][tn] = __builtin_amdgcn_wmma_f32_16x16x32_bf16(
            false, a, false, b, (short)0, acc[tm][tn], false, false);
      }
    }
    __syncthreads();
  }

  // ---- store C (C layout: vgpr r -> row sub*8+r, col = lr) ----
#pragma unroll
  for (int tm = 0; tm < 2; ++tm) {
#pragma unroll
    for (int tn = 0; tn < 2; ++tn) {
      const int col = n0 + wn * 32 + tn * 16 + lr;
      if (col < N) {
#pragma unroll
        for (int r = 0; r < 8; ++r) {
          const int row = m0 + wm * 32 + tm * 16 + sub * 8 + r;
          C[(size_t)row * N + col] = acc[tm][tn][r];
        }
      }
    }
  }
}

// ---------------------------------------------------------------------------
// RoPE on q/k slices of fused, pack bf16 q/k (b,h,t,d) and v^T (b,h,d,t).
// ---------------------------------------------------------------------------
__global__ __launch_bounds__(256) void rope_pack(
    const float* __restrict__ fused,
    __bf16* __restrict__ qb, __bf16* __restrict__ kb, __bf16* __restrict__ vtb)
{
  const size_t idx = (size_t)blockIdx.x * 256 + threadIdx.x;
  if (idx >= (size_t)B_ * T_ * NH_ * HD_) return;
  const int d = (int)(idx & 127);
  const int h = (int)((idx >> 7) & 7);
  const int t = (int)((idx >> 10) & 2047);
  const int b = (int)(idx >> 21);

  const float* row = fused + ((size_t)b * T_ + t) * FUSED_;

  const int i = d & 63;
  const float inv = __powf(10000.0f, -(float)i * (1.0f / 64.0f));
  float s, c;
  __sincosf((float)t * inv, &s, &c);

  const int dp  = (d < 64) ? d + 64 : d - 64;
  const float sg = (d < 64) ? -1.0f : 1.0f;

  const float q0 = row[OFF_Q + h * HD_ + d];
  const float qp = row[OFF_Q + h * HD_ + dp];
  const float k0 = row[OFF_K + h * HD_ + d];
  const float kp = row[OFF_K + h * HD_ + dp];

  const size_t ob = (((size_t)(b * NH_ + h)) * T_ + t) * HD_ + d;
  qb[ob] = (__bf16)(q0 * c + sg * qp * s);
  kb[ob] = (__bf16)(k0 * c + sg * kp * s);
  vtb[(((size_t)(b * NH_ + h)) * HD_ + d) * T_ + t] = (__bf16)row[OFF_V + h * HD_ + d];
}

// ---------------------------------------------------------------------------
// Flash attention: block = 128 query rows, 8 waves x 16 rows each.
// Q fragments live in registers across the whole key loop.
// K and V^T tiles are DMA'd into double-buffered LDS by the Tensor Data
// Mover (wave 0 issues; TDM ops of one wave complete in order so
// s_wait_tensorcnt(2) isolates the previous pair while the next pair flies).
// ---------------------------------------------------------------------------
#define A_LDK 136   // K tile   [64 x 128] bf16: 64 dwords/row + 4 dword pad
#define A_LDV 72    // V^T tile [128 x 64] bf16: 32 dwords/row + 4 dword pad
#define A_LDP 72    // P tile   [128 x 64] bf16, padded

__global__ __launch_bounds__(256) void flash_attn(
    const __bf16* __restrict__ qb, const __bf16* __restrict__ kb,
    const __bf16* __restrict__ vtb, float* __restrict__ cc)
{
  __shared__ __bf16 Kt [2][64  * A_LDK];
  __shared__ __bf16 Vts[2][128 * A_LDV];
  __shared__ __bf16 Ps [128 * A_LDP];

  const int tid  = threadIdx.x;
  const int lane = tid & 31;
  const int w    = tid >> 5;        // wave 0..7 -> rows w*16..w*16+15
  const int sub  = lane >> 4;
  const int lr   = lane & 15;
  const int bh   = blockIdx.y;      // b*NH + h
  const int b    = bh >> 3;
  const int h    = bh & 7;
  const int qblk = blockIdx.x;      // 0..15 (128 rows each)
  const int qrow0 = qblk * 128 + w * 16;

  const unsigned kt_lds[2]  = { (unsigned)(uintptr_t)(void*)&Kt[0][0],
                                (unsigned)(uintptr_t)(void*)&Kt[1][0] };
  const unsigned vts_lds[2] = { (unsigned)(uintptr_t)(void*)&Vts[0][0],
                                (unsigned)(uintptr_t)(void*)&Vts[1][0] };

  const __bf16* kbase = kb  + ((size_t)bh) * T_ * HD_;
  const __bf16* vbase = vtb + ((size_t)bh) * HD_ * T_;

  // preload Q fragments (A-operand layout) into registers
  bf16x16 qf[4];
  {
    const __bf16* qp = qb + (((size_t)bh) * T_ + (qrow0 + lr)) * HD_;
#pragma unroll
    for (int cch = 0; cch < 4; ++cch) {
      const bf16x8 lo = *(const bf16x8*)(qp + cch * 32 + sub * 8);
      const bf16x8 hi = *(const bf16x8*)(qp + cch * 32 + sub * 8 + 16);
      qf[cch] = bf_combine(lo, hi);
    }
  }

  v8f of[8] = {};
  float m_r[8], l_r[8];
#pragma unroll
  for (int r = 0; r < 8; ++r) { m_r[r] = -3.0e38f; l_r[r] = 0.0f; }

  const float scale = 0.08838834764831845f;  // 1/sqrt(128)
  const int jmax = 2 * qblk + 1;

  // prime the pipeline: DMA tile j=0 into buffer 0
  if (w == 0) {
    tdm_load_2d_bf16(kt_lds[0],  kbase, HD_, HD_, 64,
                     (unsigned long long)HD_, /*pad_i=*/5, /*pad_a=*/3);
    tdm_load_2d_bf16(vts_lds[0], vbase, T_, 64, HD_,
                     (unsigned long long)T_,  /*pad_i=*/4, /*pad_a=*/3);
  }

  for (int j = 0; j <= jmax; ++j) {
    const int cur = j & 1;
    // ---- issue next tile's DMA, then wait for current tile only ----
    if (w == 0) {
      if (j < jmax) {
        const int nxt = (j + 1) & 1;
        tdm_load_2d_bf16(kt_lds[nxt],  kbase + (size_t)(j + 1) * 64 * HD_,
                         HD_, HD_, 64, (unsigned long long)HD_, 5, 3);
        tdm_load_2d_bf16(vts_lds[nxt], vbase + (j + 1) * 64,
                         T_, 64, HD_, (unsigned long long)T_, 4, 3);
        __builtin_amdgcn_s_wait_tensorcnt(2);   // j's pair done, j+1 in flight
      } else {
        __builtin_amdgcn_s_wait_tensorcnt(0);
      }
    }
    __syncthreads();

    // ---- S = Q K^T : 4 col tiles x 4 K-chunks ----
    v8f sf[4] = {};
#pragma unroll
    for (int cch = 0; cch < 4; ++cch) {
#pragma unroll
      for (int n = 0; n < 4; ++n) {
        const bf16x8 blo = *(const bf16x8*)&Kt[cur][(n * 16 + lr) * A_LDK + cch * 32 + sub * 8];
        const bf16x8 bhi = *(const bf16x8*)&Kt[cur][(n * 16 + lr) * A_LDK + cch * 32 + sub * 8 + 16];
        const bf16x16 bb = bf_combine(blo, bhi);
        sf[n] = __builtin_amdgcn_wmma_f32_16x16x32_bf16(
            false, qf[cch], false, bb, (short)0, sf[n], false, false);
      }
    }

    // ---- scale + causal mask ----
    float sv[4][8];
#pragma unroll
    for (int n = 0; n < 4; ++n) {
      const int tcol = j * 64 + n * 16 + lr;
#pragma unroll
      for (int r = 0; r < 8; ++r) {
        float s = sf[n][r] * scale;
        const int trow = qrow0 + sub * 8 + r;
        if (tcol > trow) s = -3.0e38f;
        sv[n][r] = s;
      }
    }

    // ---- streaming softmax (row reductions across 16 lanes) ----
#pragma unroll
    for (int r = 0; r < 8; ++r) {
      float mx = fmaxf(fmaxf(sv[0][r], sv[1][r]), fmaxf(sv[2][r], sv[3][r]));
#pragma unroll
      for (int off = 8; off; off >>= 1) mx = fmaxf(mx, __shfl_xor(mx, off, 32));
      const float mn  = fmaxf(m_r[r], mx);
      const float cor = __expf(m_r[r] - mn);
      m_r[r] = mn;
      float rs = 0.f;
#pragma unroll
      for (int n = 0; n < 4; ++n) {
        const float p = __expf(sv[n][r] - mn);
        rs += p;
        Ps[(w * 16 + sub * 8 + r) * A_LDP + n * 16 + lr] = (__bf16)p;
      }
#pragma unroll
      for (int off = 8; off; off >>= 1) rs += __shfl_xor(rs, off, 32);
      l_r[r] = l_r[r] * cor + rs;
#pragma unroll
      for (int n = 0; n < 8; ++n) of[n][r] = of[n][r] * cor;
    }
    __syncthreads();

    // ---- O += P V : 8 col tiles x 2 K-chunks ----
#pragma unroll
    for (int cch = 0; cch < 2; ++cch) {
      const bf16x8 alo = *(const bf16x8*)&Ps[(w * 16 + lr) * A_LDP + cch * 32 + sub * 8];
      const bf16x8 ahi = *(const bf16x8*)&Ps[(w * 16 + lr) * A_LDP + cch * 32 + sub * 8 + 16];
      const bf16x16 aa = bf_combine(alo, ahi);
#pragma unroll
      for (int n = 0; n < 8; ++n) {
        const bf16x8 blo = *(const bf16x8*)&Vts[cur][(n * 16 + lr) * A_LDV + cch * 32 + sub * 8];
        const bf16x8 bhi = *(const bf16x8*)&Vts[cur][(n * 16 + lr) * A_LDV + cch * 32 + sub * 8 + 16];
        const bf16x16 bb = bf_combine(blo, bhi);
        of[n] = __builtin_amdgcn_wmma_f32_16x16x32_bf16(
            false, aa, false, bb, (short)0, of[n], false, false);
      }
    }
    __syncthreads();   // all waves done with buf[cur] before it is re-DMA'd
  }

  // ---- normalize and store into concat[:, h*128 + col] ----
#pragma unroll
  for (int n = 0; n < 8; ++n) {
    const int col = h * HD_ + n * 16 + lr;
#pragma unroll
    for (int r = 0; r < 8; ++r) {
      const int trow = qrow0 + sub * 8 + r;
      cc[((size_t)b * T_ + trow) * 2048 + col] = of[n][r] / l_r[r];
    }
  }
}

// ---------------------------------------------------------------------------
// Whiteboard scan: 1 block per (b, head), 128 lanes, lane v owns S[:, v].
// ---------------------------------------------------------------------------
__global__ __launch_bounds__(128) void wb_scan(
    const float* __restrict__ fused, const float* __restrict__ S0,
    const float* __restrict__ dbias, const float* __restrict__ demb,
    const float* __restrict__ nw, float* __restrict__ cc,
    float* __restrict__ Sout)
{
  const int bh = blockIdx.x;
  const int b = bh >> 3, h = bh & 7;
  const int v = threadIdx.x;

  __shared__ float kbuf[WK_], qbuf[WK_], sc[2], red[WV_];

  float S[WK_];
#pragma unroll
  for (int k = 0; k < WK_; ++k)
    S[k] = S0[(((size_t)bh) * WK_ + k) * WV_ + v];
  const float nwv = nw[v];

  for (int t = 0; t < T_; ++t) {
    const float* row = fused + ((size_t)b * T_ + t) * FUSED_;
    if (v < WK_) {
      kbuf[v] = row[OFF_KW + h * WK_ + v] + demb[h * WK_ + v];
      qbuf[v] = row[OFF_QW + h * WK_ + v] * 0.125f;   // WK^-0.5
    }
    if (v == 0) {
      sc[0] = sigmoidf_(row[OFF_BETA + h]);                 // beta
      sc[1] = sigmoidf_(row[OFF_DEC + h] + dbias[h]);       // exp(log_sigmoid)
    }
    __syncthreads();

    const float beta = sc[0], dec = sc[1];
    float dot = 0.f;
#pragma unroll
    for (int k = 0; k < WK_; ++k) { S[k] *= dec; dot = fmaf(kbuf[k], S[k], dot); }
    const float vt   = row[OFF_VW + h * WV_ + v];
    const float vnew = (vt - dot) * beta;
    float o = 0.f;
#pragma unroll
    for (int k = 0; k < WK_; ++k) { S[k] = fmaf(kbuf[k], vnew, S[k]); o = fmaf(qbuf[k], S[k], o); }

    red[v] = o * o;
    __syncthreads();
    for (int s2 = 64; s2 > 0; s2 >>= 1) {
      if (v < s2) red[v] += red[v + s2];
      __syncthreads();
    }
    const float var  = red[0] * (1.0f / (float)WV_);
    const float gate = sigmoidf_(row[OFF_GWB + h * WV_ + v]);
    const float outv = nwv * o * rsqrtf(var + 1e-5f) * gate;
    cc[((size_t)b * T_ + t) * 2048 + 1024 + h * WV_ + v] = outv;
    __syncthreads();  // protect kbuf/sc/red for next step
  }

#pragma unroll
  for (int k = 0; k < WK_; ++k)
    Sout[(((size_t)bh) * WK_ + k) * WV_ + v] = S[k];
}

// ---------------------------------------------------------------------------
// Launch
// ---------------------------------------------------------------------------
extern "C" void kernel_launch(void* const* d_in, const int* in_sizes, int n_in,
                              void* d_out, int out_size, void* d_ws, size_t ws_size,
                              hipStream_t stream) {
  const float* X     = (const float*)d_in[0];  // hidden_states (B,T,HID)
  const float* S0    = (const float*)d_in[1];  // whiteboard_state
  const float* Wqkv  = (const float*)d_in[2];  // (HID, FUSED)
  const float* dbias = (const float*)d_in[3];  // (WH,)
  const float* demb  = (const float*)d_in[4];  // (1,1,WH,WK)
  const float* nw    = (const float*)d_in[5];  // (WV,)
  const float* Wo    = (const float*)d_in[6];  // (2048, HID)
  float* out = (float*)d_out;

  char* ws = (char*)d_ws;
  const size_t M = (size_t)B_ * T_;                 // 4096
  float*  fused = (float*)ws;                        ws += M * FUSED_ * sizeof(float);
  __bf16* qb    = (__bf16*)ws;                       ws += M * HID_ * sizeof(__bf16);
  __bf16* kb    = (__bf16*)ws;                       ws += M * HID_ * sizeof(__bf16);
  __bf16* vtb   = (__bf16*)ws;                       ws += M * HID_ * sizeof(__bf16);
  float*  cc    = (float*)ws;                        ws += M * 2048 * sizeof(float);

  // 1) fused = X @ W_qkv            (4096 x 6160, K=1024)
  gemm_bf16_wmma<<<dim3((FUSED_ + 127) / 128, (int)(M / 64)), 256, 0, stream>>>(
      X, Wqkv, fused, (int)M, FUSED_, HID_);

  // 2) RoPE + bf16 pack of q, k, v^T
  rope_pack<<<(int)((M * HID_) / 256), 256, 0, stream>>>(fused, qb, kb, vtb);

  // 3) flash attention -> concat[:, 0:1024]
  flash_attn<<<dim3(T_ / 128, B_ * NH_), 256, 0, stream>>>(qb, kb, vtb, cc);

  // 4) whiteboard scan -> concat[:, 1024:2048], S_final -> tail of d_out
  wb_scan<<<B_ * WH_, WV_, 0, stream>>>(
      fused, S0, dbias, demb, nw, cc, out + M * HID_);

  // 5) out = concat @ W_o           (4096 x 1024, K=2048)
  gemm_bf16_wmma<<<dim3(HID_ / 128, (int)(M / 64)), 256, 0, stream>>>(
      cc, Wo, out, (int)M, HID_, 2048);
}
